// VGG_SNN_STDB_62483184222553
// MI455X (gfx1250) — matrix operand order repaired and checked by reading, not compiled, not run on hardware
//
#include <hip/hip_runtime.h>
#include <hip/hip_bf16.h>

typedef __bf16 v16bf __attribute__((ext_vector_type(16)));
typedef float  v8f   __attribute__((ext_vector_type(8)));

#define LEAK 0.99f

#if __has_builtin(__builtin_amdgcn_global_load_async_to_lds_b128) && \
    __has_builtin(__builtin_amdgcn_s_wait_asynccnt)
#define HAVE_ASYNC_LDS 1
typedef int v4i_async __attribute__((vector_size(16)));
typedef __attribute__((address_space(1))) v4i_async* gv4p;
typedef __attribute__((address_space(3))) v4i_async* lv4p;
#else
#define HAVE_ASYNC_LDS 0
#endif

// ---------------- layer tables ----------------
static const int CONV_CI [13] = {  3, 64, 64,128,128,256,256,256,512,512,512,512,512};
static const int CONV_CIP[13] = { 32, 64, 64,128,128,256,256,256,512,512,512,512,512};
static const int CONV_CO [13] = { 64, 64,128,128,256,256,256,512,512,512,512,512,512};
static const int CONV_S  [13] = {224,224,112,112, 56, 56, 56, 28, 28, 28, 14, 14, 14};
static const int POST[13] = {1,2,1,2,1,1,2,1,1,2,1,1,2};   // 1=dropout, 2=avgpool2
static const int DIDX[13] = {0,-1,1,-1,2,3,-1,4,5,-1,6,7,-1};

// ---------------- weight repack into per-lane WMMA A fragments, contiguous k = 16*(lane/16) + j ----------------
__global__ __launch_bounds__(256) void repack_w(const float* __restrict__ w,
                                                __bf16* __restrict__ out,
                                                int Cout, int Cin, int Cinp) {
    int Kt = Cinp >> 5;
    long total = (long)Cout * Cinp * 9;
    long e = blockIdx.x * (long)blockDim.x + threadIdx.x;
    if (e >= total) return;
    int  j    = (int)(e & 15);
    long r    = e >> 4;
    int  lane = (int)(r & 31);  r >>= 5;
    int  kt   = (int)(r % Kt);  r /= Kt;
    int  s    = (int)(r % 9);
    int  mt   = (int)(r / 9);
    int  m    = mt * 16 + (lane & 15);
    int  khi  = (lane >> 4) << 4;
    int  kin  = j + khi;
    int  k    = kt * 32 + kin;
    float v = 0.0f;
    if (k < Cin) v = w[((long)m * Cin + k) * 9 + s];
    out[e] = (__bf16)v;
}

// ---------------- Poisson spike generator -> padded HWC bf16 (C=32) ----------------
__global__ __launch_bounds__(256) void poisson_spk(const float* __restrict__ x,
                                                   __bf16* __restrict__ spk, int t) {
    long n = 226L * 226L * 32L;
    long idx = blockIdx.x * (long)blockDim.x + threadIdx.x;
    if (idx >= n) return;
    int c = (int)(idx % 32);
    long r = idx / 32;
    int xx = (int)(r % 226), yy = (int)(r / 226);
    float v = 0.0f;
    if (c < 3 && xx >= 1 && xx <= 224 && yy >= 1 && yy <= 224) {
        long pi = (long)c * 224 * 224 + (long)(yy - 1) * 224 + (xx - 1);
        unsigned u = (unsigned)pi * 2654435761u + (unsigned)(t + 1) * 0x9E3779B9u;
        u ^= u >> 16; u *= 0x7feb352du; u ^= u >> 15; u *= 0x846ca68bu; u ^= u >> 16;
        float rr = (float)(u >> 8) * (1.0f / 16777216.0f);
        v = (rr <= x[pi]) ? 1.0f : 0.0f;
    }
    spk[idx] = (__bf16)v;
}

// ============ direct conv (small Mt layers 0,1): global B loads ============
template<int S, int CIP, int CO>
__global__ __launch_bounds__(256) void conv_direct(const __bf16* __restrict__ spk,
                                                   const __bf16* __restrict__ wA,
                                                   float* __restrict__ out) {
    constexpr int Mt = CO >> 4, Npix = S * S, PG = (Npix + 63) >> 6;
    constexpr int Kt = CIP >> 5, Sp = S + 2;
    int wid  = blockIdx.x * 8 + (threadIdx.x >> 5);
    int lane = threadIdx.x & 31;
    if (wid >= Mt * PG) return;                   // wave-uniform
    int mt = wid % Mt, pixbase = (wid / Mt) * 64;
    int nl = lane & 15;
    int px[4], py[4]; bool valid[4];
#pragma unroll
    for (int g = 0; g < 4; ++g) {
        int p = pixbase + g * 16 + nl;
        valid[g] = (p < Npix);
        int pc = valid[g] ? p : (Npix - 1);
        px[g] = pc % S; py[g] = pc / S;
    }
    int koff = (lane >> 4) << 4;
    v8f acc[4] = {};
#pragma unroll
    for (int s = 0; s < 9; ++s) {
        int dy = s / 3, dx = s % 3;
        const __bf16* wp = wA + (((long)mt * 9 + s) * Kt << 9) + lane * 16;
        const __bf16* bp[4];
#pragma unroll
        for (int g = 0; g < 4; ++g)
            bp[g] = spk + ((long)(py[g] + dy) * Sp + (px[g] + dx)) * CIP + koff;
#pragma unroll
        for (int kc = 0; kc < Kt; ++kc) {
            v16bf a = *(const v16bf*)(wp + (kc << 9));
#pragma unroll
            for (int g = 0; g < 4; ++g) {
                v16bf b = *(const v16bf*)(bp[g] + (kc << 5));
                acc[g] = __builtin_amdgcn_wmma_f32_16x16x32_bf16(
                    false, a, false, b, (short)0, acc[g], false, false);
            }
        }
    }
    int mrow = (lane >> 4) << 3, cobase = mt << 4;
#pragma unroll
    for (int g = 0; g < 4; ++g) {
        if (!valid[g]) continue;
        int p = pixbase + g * 16 + nl;
        float* op = out + (long)(cobase + mrow) * Npix + p;
#pragma unroll
        for (int r = 0; r < 8; ++r) op[(long)r * Npix] = acc[g][r];
    }
}

// ============ LDS-staged conv (Mt multiple of 8): async global->LDS B tiles ============
// 8 waves/block share one 64-pixel group; K-chunks (32ch) double-buffered in LDS.
template<int S, int CIP, int CO>
__global__ __launch_bounds__(256) void conv_lds(const __bf16* __restrict__ spk,
                                                const __bf16* __restrict__ wA,
                                                float* __restrict__ out) {
    constexpr int Mt = CO >> 4, Npix = S * S, PG = (Npix + 63) >> 6;
    constexpr int Kt = CIP >> 5, Sp = S + 2;
    constexpr int LROWS  = (S + 62) / S + 3;     // row span of 64px group + 2 halo rows
    constexpr int TILE_H = LROWS * Sp * 32;      // bf16 elements per K-chunk tile
    constexpr int PIECES = LROWS * Sp * 4;       // 16-byte pieces per tile

    __shared__ __bf16 buf[2][TILE_H];

    int tid  = threadIdx.x;
    int wib  = tid >> 5;
    int lane = tid & 31;
    int pg   = blockIdx.x % PG;
    int mt   = (blockIdx.x / PG) * 8 + wib;
    int pixbase = pg * 64;
    int y0 = pixbase / S;
    int nl = lane & 15;
    int koff = (lane >> 4) << 4;

    int lx[4], ly[4]; bool valid[4];
#pragma unroll
    for (int g = 0; g < 4; ++g) {
        int p = pixbase + g * 16 + nl;
        valid[g] = (p < Npix);
        int pc = valid[g] ? p : (Npix - 1);
        ly[g] = pc / S - y0;
        lx[g] = pc % S;
    }

    const __bf16* wp = wA + ((long)mt * 9 * Kt << 9) + lane * 16;

    auto stage = [&](int kc, __bf16* dst) {
        for (int i = tid; i < PIECES; i += 256) {
            int row = i / (Sp * 4);
            int rem = i - row * (Sp * 4);
            int px  = rem >> 2;
            int q   = rem & 3;
            int rs  = y0 + row; if (rs > S + 1) rs = S + 1;   // clamp (only feeds invalid px)
            const __bf16* src = spk + ((long)rs * Sp + px) * CIP + kc * 32 + q * 8;
            __bf16* d = dst + i * 8;
#if HAVE_ASYNC_LDS
            __builtin_amdgcn_global_load_async_to_lds_b128((gv4p)src, (lv4p)d, 0, 0);
#else
            *(int4*)d = *(const int4*)src;
#endif
        }
    };

    v8f acc[4] = {};
    stage(0, buf[0]);
    for (int kc = 0; kc < Kt; ++kc) {
#if HAVE_ASYNC_LDS
        __builtin_amdgcn_s_wait_asynccnt(0);     // this wave's async fills landed
#endif
        __syncthreads();                          // everyone's fills visible, prior reads done
        if (kc + 1 < Kt) stage(kc + 1, buf[(kc + 1) & 1]);
        const __bf16* bb = buf[kc & 1];
#pragma unroll
        for (int s = 0; s < 9; ++s) {
            int dy = s / 3, dx = s % 3;
            v16bf a = *(const v16bf*)(wp + ((s * Kt + kc) << 9));
#pragma unroll
            for (int g = 0; g < 4; ++g) {
                v16bf b = *(const v16bf*)(bb + (((ly[g] + dy) * Sp + lx[g] + dx) << 5) + koff);
                acc[g] = __builtin_amdgcn_wmma_f32_16x16x32_bf16(
                    false, a, false, b, (short)0, acc[g], false, false);
            }
        }
    }
    int mrow = (lane >> 4) << 3, cobase = mt << 4;
#pragma unroll
    for (int g = 0; g < 4; ++g) {
        if (!valid[g]) continue;
        int p = pixbase + g * 16 + nl;
        float* op = out + (long)(cobase + mrow) * Npix + p;
#pragma unroll
        for (int r = 0; r < 8; ++r) op[(long)r * Npix] = acc[g][r];
    }
}

// ---------------- LIF conv (+dropout) -> next padded HWC spike map ----------------
__global__ __launch_bounds__(256) void lif_conv(float* __restrict__ mem,
                                                const float* __restrict__ conv,
                                                __bf16* __restrict__ spkout,
                                                const float* __restrict__ mask,
                                                const float* __restrict__ thrs, int ti,
                                                int C, int S) {
    long n = (long)C * S * S;
    long idx = blockIdx.x * (long)blockDim.x + threadIdx.x;
    if (idx >= n) return;
    float thr = thrs[ti];
    int Npix = S * S;
    int c = (int)(idx / Npix), p = (int)(idx % Npix);
    float m = mem[idx];
    float spike = (m / thr - 1.0f > 0.0f) ? 1.0f : 0.0f;
    mem[idx] = LEAK * m + conv[idx] - thr * spike;
    float v = spike;
    if (mask) v *= mask[idx];
    int y = p / S, xx = p % S;
    spkout[((long)(y + 1) * (S + 2) + (xx + 1)) * C + c] = (__bf16)v;
}

// ---------------- 2x2 avgpool on padded HWC spike maps ----------------
__global__ __launch_bounds__(256) void pool2(const __bf16* __restrict__ in,
                                             __bf16* __restrict__ out, int C, int S) {
    int So = S >> 1;
    long n = (long)C * So * So;
    long idx = blockIdx.x * (long)blockDim.x + threadIdx.x;
    if (idx >= n) return;
    int c = (int)(idx % C);
    long r = idx / C;
    int xo = (int)(r % So), yo = (int)(r / So);
    float v = 0.0f;
#pragma unroll
    for (int dy = 0; dy < 2; ++dy)
#pragma unroll
        for (int dx = 0; dx < 2; ++dx)
            v += (float)in[((long)(2 * yo + dy + 1) * (S + 2) + (2 * xo + dx + 1)) * C + c];
    out[((long)(yo + 1) * (So + 2) + (xo + 1)) * C + c] = (__bf16)(0.25f * v);
}

// ---------------- flatten pooled 7x7x512 (CHW order) -> f32 vector ----------------
__global__ __launch_bounds__(256) void flatten7(const __bf16* __restrict__ pooled,
                                                float* __restrict__ fcin) {
    int k = blockIdx.x * blockDim.x + threadIdx.x;
    if (k >= 25088) return;
    int c = k / 49, r = k % 49, y = r / 7, x = r % 7;
    fcin[k] = (float)pooled[((long)(y + 1) * 9 + (x + 1)) * 512 + c];
}

// ---------------- FC GEMV: one wave per row, b128 streaming ----------------
__global__ __launch_bounds__(256) void fc_gemv(const float* __restrict__ w,
                                               const float* __restrict__ x,
                                               float* __restrict__ y,
                                               int R, int K, int accumulate) {
    int wid  = blockIdx.x * 8 + (threadIdx.x >> 5);
    int lane = threadIdx.x & 31;
    if (wid >= R) return;
    const float4* wr4 = (const float4*)(w + (long)wid * K);
    const float4* x4  = (const float4*)x;
    int K4 = K >> 2;
    float s = 0.0f;
    for (int k = lane; k < K4; k += 32) {
        __builtin_prefetch(wr4 + k + 256, 0, 0);
        float4 a = wr4[k], b = x4[k];
        s += a.x * b.x + a.y * b.y + a.z * b.z + a.w * b.w;
    }
#pragma unroll
    for (int off = 16; off; off >>= 1) s += __shfl_xor(s, off, 32);
    if (lane == 0) y[wid] = accumulate ? (y[wid] + s) : s;
}

// ---------------- LIF for FC layers ----------------
__global__ __launch_bounds__(256) void lif_fc(float* __restrict__ mem,
                                              const float* __restrict__ y,
                                              const float* __restrict__ mask,
                                              const float* __restrict__ thrs, int ti,
                                              float* __restrict__ outv, int n) {
    int i = blockIdx.x * blockDim.x + threadIdx.x;
    if (i >= n) return;
    float thr = thrs[ti];
    float m = mem[i];
    float spike = (m / thr - 1.0f > 0.0f) ? 1.0f : 0.0f;
    mem[i] = LEAK * m + y[i] - thr * spike;
    outv[i] = spike * mask[i];
}

// ---------------- host orchestration ----------------
static inline int cdiv(long a, long b) { return (int)((a + b - 1) / b); }

static void conv_dispatch(int i, const __bf16* spk, const __bf16* wA, float* out,
                          hipStream_t st) {
#define CASE_DIR(I, S_, CIP_, CO_)                                                    \
    case I: {                                                                         \
        int waves = ((CO_) >> 4) * (((S_) * (S_) + 63) >> 6);                         \
        conv_direct<S_, CIP_, CO_><<<(waves + 7) / 8, 256, 0, st>>>(spk, wA, out);    \
    } break;
#define CASE_LDS(I, S_, CIP_, CO_)                                                    \
    case I: {                                                                         \
        int blocks = (((CO_) >> 4) / 8) * (((S_) * (S_) + 63) >> 6);                  \
        conv_lds<S_, CIP_, CO_><<<blocks, 256, 0, st>>>(spk, wA, out);                \
    } break;
    switch (i) {
        CASE_DIR(0, 224, 32, 64)
        CASE_DIR(1, 224, 64, 64)
        CASE_LDS(2, 112, 64, 128)
        CASE_LDS(3, 112, 128, 128)
        CASE_LDS(4, 56, 128, 256)
        CASE_LDS(5, 56, 256, 256)
        CASE_LDS(6, 56, 256, 256)
        CASE_LDS(7, 28, 256, 512)
        CASE_LDS(8, 28, 512, 512)
        CASE_LDS(9, 28, 512, 512)
        CASE_LDS(10, 14, 512, 512)
        CASE_LDS(11, 14, 512, 512)
        CASE_LDS(12, 14, 512, 512)
        default: break;
    }
#undef CASE_DIR
#undef CASE_LDS
}

extern "C" void kernel_launch(void* const* d_in, const int* in_sizes, int n_in,
                              void* d_out, int out_size, void* d_ws, size_t ws_size,
                              hipStream_t stream) {
    (void)in_sizes; (void)n_in; (void)ws_size;
    const float* x = (const float*)d_in[0];
    const float* conv_w[13]; for (int i = 0; i < 13; ++i) conv_w[i] = (const float*)d_in[1 + i];
    const float* fc_w[3];    for (int i = 0; i < 3;  ++i) fc_w[i]   = (const float*)d_in[14 + i];
    const float* mfeat[8];   for (int i = 0; i < 8;  ++i) mfeat[i]  = (const float*)d_in[17 + i];
    const float* mfc[2];     for (int i = 0; i < 2;  ++i) mfc[i]    = (const float*)d_in[25 + i];
    const float* thrs = (const float*)d_in[27];
    const int T = 4;

    char* ws = (char*)d_ws;
    size_t off = 0;
    auto alloc = [&](size_t bytes) -> void* {
        void* p = ws + off;
        off = (off + bytes + 255) & ~(size_t)255;
        return p;
    };

    __bf16* wRep[13];
    for (int i = 0; i < 13; ++i)
        wRep[i] = (__bf16*)alloc((size_t)CONV_CO[i] * CONV_CIP[i] * 9 * sizeof(__bf16));
    float* mem_c[13];
    for (int i = 0; i < 13; ++i)
        mem_c[i] = (float*)alloc((size_t)CONV_CO[i] * CONV_S[i] * CONV_S[i] * sizeof(float));
    float*  convOut = (float*)alloc((size_t)64 * 224 * 224 * sizeof(float));
    __bf16* spkbuf[3];
    for (int i = 0; i < 3; ++i)
        spkbuf[i] = (__bf16*)alloc((size_t)226 * 226 * 64 * sizeof(__bf16));
    float* fc_in  = (float*)alloc(25088 * sizeof(float));
    float* fc_buf = (float*)alloc(4096 * sizeof(float));
    float* fc_act = (float*)alloc(4096 * sizeof(float));
    float* mem_f0 = (float*)alloc(4096 * sizeof(float));
    float* mem_f1 = (float*)alloc(4096 * sizeof(float));

    (void)hipMemsetAsync(d_out, 0, (size_t)out_size * sizeof(float), stream);
    for (int i = 0; i < 13; ++i)
        (void)hipMemsetAsync(mem_c[i], 0, (size_t)CONV_CO[i] * CONV_S[i] * CONV_S[i] * sizeof(float), stream);
    (void)hipMemsetAsync(mem_f0, 0, 4096 * sizeof(float), stream);
    (void)hipMemsetAsync(mem_f1, 0, 4096 * sizeof(float), stream);

    for (int i = 0; i < 13; ++i) {
        long tot = (long)CONV_CO[i] * CONV_CIP[i] * 9;
        repack_w<<<cdiv(tot, 256), 256, 0, stream>>>(conv_w[i], wRep[i],
                                                     CONV_CO[i], CONV_CI[i], CONV_CIP[i]);
    }

    int slot = 0;
    auto nextbuf = [&]() -> __bf16* { __bf16* p = spkbuf[slot]; slot = (slot + 1) % 3; return p; };

    for (int t = 0; t < T; ++t) {
        __bf16* cur = nextbuf();
        poisson_spk<<<cdiv(226L * 226 * 32, 256), 256, 0, stream>>>(x, cur, t);

        for (int i = 0; i < 13; ++i) {
            int S = CONV_S[i], Co = CONV_CO[i];
            conv_dispatch(i, cur, wRep[i], convOut, stream);
            __bf16* nxt = nextbuf();
            (void)hipMemsetAsync(nxt, 0, (size_t)(S + 2) * (S + 2) * Co * sizeof(__bf16), stream);
            const float* mask = (DIDX[i] >= 0) ? mfeat[DIDX[i]] : nullptr;
            long nel = (long)Co * S * S;
            lif_conv<<<cdiv(nel, 256), 256, 0, stream>>>(mem_c[i], convOut, nxt, mask,
                                                         thrs, i, Co, S);
            cur = nxt;
            if (POST[i] == 2) {
                int So = S >> 1;
                __bf16* pl = nextbuf();
                (void)hipMemsetAsync(pl, 0, (size_t)(So + 2) * (So + 2) * Co * sizeof(__bf16), stream);
                long np = (long)Co * So * So;
                pool2<<<cdiv(np, 256), 256, 0, stream>>>(cur, pl, Co, S);
                cur = pl;
            }
        }

        flatten7<<<cdiv(25088, 256), 256, 0, stream>>>(cur, fc_in);
        fc_gemv<<<cdiv(4096, 8), 256, 0, stream>>>(fc_w[0], fc_in, fc_buf, 4096, 25088, 0);
        lif_fc <<<cdiv(4096, 256), 256, 0, stream>>>(mem_f0, fc_buf, mfc[0], thrs, 13, fc_act, 4096);
        fc_gemv<<<cdiv(4096, 8), 256, 0, stream>>>(fc_w[1], fc_act, fc_buf, 4096, 4096, 0);
        lif_fc <<<cdiv(4096, 256), 256, 0, stream>>>(mem_f1, fc_buf, mfc[1], thrs, 14, fc_act, 4096);
        fc_gemv<<<cdiv(1000, 8), 256, 0, stream>>>(fc_w[2], fc_act, (float*)d_out, 1000, 4096, 1);
    }
}
